// ct_layer_36344013258766
// MI455X (gfx1250) — matrix operand order repaired and checked by reading, not compiled, not run on hardware
//
#include <hip/hip_runtime.h>

// ---------------------------------------------------------------------------
// CornerNet-style corner-pool layer for MI455X (gfx1250).
// B=8, C=256, H=W=96, five 3x3 convs as implicit GEMM on bf16 WMMA.
// Round 2: 32-pixel M tile (B-fragment reuse x2 -> 32 FLOP/B from L2),
//          async global->LDS patch staging (ASYNCcnt path).
// ---------------------------------------------------------------------------

#define B_ 8
#define C_ 256
#define H_ 96
#define W_ 96

typedef __attribute__((ext_vector_type(16))) __bf16       v16bf;
typedef __attribute__((ext_vector_type(8)))  float        v8f;
typedef __attribute__((ext_vector_type(4)))  unsigned int u4;
typedef __attribute__((ext_vector_type(4)))  float        f4;

static __device__ __forceinline__ unsigned short f2bf(float f) {
  union { float f; unsigned u; } v; v.f = f;
  unsigned r = v.u + 0x7FFFu + ((v.u >> 16) & 1u);   // round-to-nearest-even
  return (unsigned short)(r >> 16);
}

// ---------------------------------------------------------------------------
// Weight prep: OIHW f32 -> fragment-swizzled bf16.
// K ordering: k = tap*256 + cin (tap = kh*3+kw). Layout:
//   wf[((kt*16 + nt)*32 + lane)*16 + e] = Wt[kt*32 + ((lane>=16)?8:0) + (e<8 ? e : 8+e)][nt*16 + (lane&15)]
// exactly the CDNA5 16-bit B-matrix (32x16) per-lane register image.
// ---------------------------------------------------------------------------
__global__ void k_prep_w(const float* __restrict__ w, unsigned short* __restrict__ wf) {
  int idx  = blockIdx.x * 256 + threadIdx.x;   // < 72*16*32*16 = 589824
  int kt   = idx >> 13;
  int rem  = idx & 8191;
  int nt   = rem >> 9;
  int rem2 = rem & 511;
  int lane = rem2 >> 4;
  int e    = rem2 & 15;
  int kk   = ((lane >> 4) << 3) + (e < 8 ? e : 8 + e);   // 0..31 within k-chunk
  int k    = kt * 32 + kk;
  int cin  = k & 255;
  int tap  = k >> 8;                           // 0..8
  int cout = nt * 16 + (lane & 15);
  float val = w[((cout * 256 + cin) * 3 + tap / 3) * 3 + (tap % 3)];
  wf[idx] = f2bf(val);
}

// ---------------------------------------------------------------------------
// NCHW f32 (optionally relu(a+b)) -> NHWC bf16, LDS-tiled transpose.
// One block per (b,h); 8 chunks of 32 channels; 97-float pitch: conflict-free.
// ---------------------------------------------------------------------------
__global__ __launch_bounds__(256) void k_nchw_to_nhwc_bf16(
    const float* __restrict__ a, const float* __restrict__ b,
    unsigned short* __restrict__ out, int add_relu) {
  __shared__ float tile[32][97];
  const int bh  = blockIdx.x;            // 0..B*H-1
  const int b0  = bh / H_;
  const int h   = bh % H_;
  const int tid = threadIdx.x;
  const size_t in_bh  = (size_t)b0 * C_ * H_ * W_ + (size_t)h * W_;
  const size_t out_bh = ((size_t)b0 * H_ + h) * W_ * C_;
  for (int cc = 0; cc < C_; cc += 32) {
    for (int it = 0; it < 12; ++it) {
      int lin = it * 256 + tid;          // 0..3071 over 32c x 96w
      int cl = lin / 96, w = lin % 96;
      size_t gi = in_bh + (size_t)(cc + cl) * (H_ * W_) + w;
      float v = a[gi];
      if (add_relu) { v += b[gi]; v = v > 0.f ? v : 0.f; }
      tile[cl][w] = v;
    }
    __syncthreads();
    for (int it = 0; it < 12; ++it) {
      int lin = it * 256 + tid;
      int w = lin / 32, cl = lin % 32;
      out[out_bh + (size_t)w * C_ + cc + cl] = f2bf(tile[cl][w]);
    }
    __syncthreads();
  }
}

// ---------------------------------------------------------------------------
// Corner pooling reductions. cummax(rev)+cummax == broadcast of full max.
// ---------------------------------------------------------------------------
__global__ void k_colmax(const float* __restrict__ p, float* __restrict__ cmax) {
  int t = blockIdx.x * 256 + threadIdx.x;        // < B*C*W
  int w  = t % W_;
  int c  = (t / W_) % C_;
  int b0 = t / (W_ * C_);
  const float* base = p + ((size_t)b0 * C_ + c) * H_ * W_ + w;
  float m = base[0];
  for (int h = 1; h < H_; ++h) { float v = base[(size_t)h * W_]; m = v > m ? v : m; }
  cmax[t] = m;
}

__global__ void k_rowmax(const float* __restrict__ p, float* __restrict__ rmax) {
  int t = blockIdx.x * 256 + threadIdx.x;        // < B*C*H
  int h  = t % H_;
  int c  = (t / H_) % C_;
  int b0 = t / (H_ * C_);
  const float* base = p + (((size_t)b0 * C_ + c) * H_ + h) * W_;
  float m = base[0];
  for (int w = 1; w < W_; ++w) { float v = base[w]; m = v > m ? v : m; }
  rmax[t] = m;
}

// s[b,h,w,c] = colmax[b,c,w] + rowmax[b,c,h]  -> NHWC bf16
__global__ void k_make_s(const float* __restrict__ cmax, const float* __restrict__ rmax,
                         unsigned short* __restrict__ sb) {
  int t = blockIdx.x * 256 + threadIdx.x;        // < B*H*W*C (c fastest == NHWC linear)
  int c    = t & 255;
  int rem  = t >> 8;
  int w    = rem % W_;
  int rem2 = rem / W_;
  int h    = rem2 % H_;
  int b0   = rem2 / H_;
  float v = cmax[(b0 * C_ + c) * W_ + w] + rmax[(b0 * C_ + c) * H_ + h];
  sb[t] = f2bf(v);
}

// ---------------------------------------------------------------------------
// 3x3 SAME conv + folded BN (+optional ReLU), implicit GEMM via
// v_wmma_f32_16x16x32_bf16. Block = 128 threads (4 waves), grid (3, 96, 8).
// Block tile: 32 output pixels (row segment) x all 256 couts.
// Wave tile : 2 M-tiles x 4 N-tiles = 8 accumulators; each B fragment is
// reused by 2 WMMAs -> 32 FLOP per byte streamed from L2 weights.
// ---------------------------------------------------------------------------
__global__ __launch_bounds__(128) void k_conv3x3_wmma(
    const unsigned short* __restrict__ xb,   // NHWC bf16 input
    const unsigned short* __restrict__ wf,   // swizzled bf16 weights
    const float* __restrict__ gamma, const float* __restrict__ beta,
    float* __restrict__ out, int do_relu) {
  // 3 rows x 34 cols x 256 cin, padded to 264 halves/row: per-m ds_b128 loads
  // step 4 banks per pixel -> minimum-conflict wide LDS reads.
  __shared__ unsigned short sx[3][34][264];

  const int tid  = threadIdx.x;
  const int lane = tid & 31;
  const int wv   = tid >> 5;
  const int w0   = blockIdx.x * 32;
  const int h    = blockIdx.y;
  const int b0   = blockIdx.z;

  // Cooperative LDS fill: 3*34*32 chunks of 8 halves (16B).
  // In-range chunks: GLOBAL_LOAD_ASYNC_TO_LDS_B128 (memory->LDS, ASYNCcnt).
  // Out-of-range chunks: zero ds_store (disjoint addresses, DScnt).
  for (int ch = tid; ch < 3264; ch += 128) {
    int r    = ch / (34 * 32);
    int rem  = ch % (34 * 32);
    int cidx = rem / 32;
    int ci8  = (rem % 32) * 8;
    int gh = h + r - 1;
    int gw = w0 + cidx - 1;
    unsigned short* lp = &sx[r][cidx][ci8];
    if ((unsigned)gh < (unsigned)H_ && (unsigned)gw < (unsigned)W_) {
      const unsigned short* gp = xb + (((size_t)b0 * H_ + gh) * W_ + gw) * C_ + ci8;
      unsigned lds_off = (unsigned)((char*)lp - (char*)&sx[0][0][0]);
      asm volatile("global_load_async_to_lds_b128 %0, %1, off"
                   :: "v"(lds_off), "v"(gp) : "memory");
    } else {
      u4 z = {0u, 0u, 0u, 0u};
      *(u4*)lp = z;
    }
  }
  asm volatile("s_wait_asynccnt 0x0" ::: "memory");
  __syncthreads();

  v8f acc[8] = {{}, {}, {}, {}, {}, {}, {}, {}};   // [mt*4 + j]
  const int nt0 = wv * 4;
  const int m   = lane & 15;               // pixel row within an M tile
  const int kb  = (lane >> 4) << 3;        // A-matrix K base: 0 or 8
  const unsigned short* wbase = wf + (size_t)nt0 * 512 + (size_t)lane * 16;

  for (int tap = 0; tap < 9; ++tap) {
    const int dh = tap / 3, dw = tap % 3;
    const unsigned short* arow0 = &sx[dh][m + dw][kb];        // pixels w0..w0+15
    const unsigned short* arow1 = &sx[dh][m + 16 + dw][kb];   // pixels w0+16..w0+31
    const unsigned short* wk    = wbase + (size_t)tap * 8 * 8192;
    if (tap < 8) __builtin_prefetch(wk + 8 * 8192, 0, 0);     // next tap's weights
#pragma unroll
    for (int cc = 0; cc < 8; ++cc) {
      // A fragments: lane holds row m, K = kb+{0..7} and kb+{16..23} of chunk.
      union { v16bf v; u4 q[2]; } A0, A1;
      const unsigned short* ap0 = arow0 + cc * 32;
      const unsigned short* ap1 = arow1 + cc * 32;
      A0.q[0] = *(const u4*)(ap0);
      A0.q[1] = *(const u4*)(ap0 + 16);
      A1.q[0] = *(const u4*)(ap1);
      A1.q[1] = *(const u4*)(ap1 + 16);
      const u4* wp = (const u4*)(wk) + cc * 1024;   // 8192 halves per k-tile
#pragma unroll
      for (int j = 0; j < 4; ++j) {
        union { v16bf v; u4 q[2]; } Bv;
        Bv.q[0] = wp[j * 64];                        // 512 halves per n-tile
        Bv.q[1] = wp[j * 64 + 1];
        acc[j] = __builtin_amdgcn_wmma_f32_16x16x32_bf16(
            false, A0.v, false, Bv.v, (short)0, acc[j], false, false);
        acc[4 + j] = __builtin_amdgcn_wmma_f32_16x16x32_bf16(
            false, A1.v, false, Bv.v, (short)0, acc[4 + j], false, false);
      }
    }
  }

  // Epilogue: BN affine (+ReLU), store f32 NCHW (8 consecutive w per lane).
  const int off = (lane >> 4) << 3;        // M = vgpr + (lane>=16 ? 8 : 0)
#pragma unroll
  for (int j = 0; j < 4; ++j) {
    const int cout = (nt0 + j) * 16 + (lane & 15);
    const float g  = gamma[cout];
    const float bt = beta[cout];
#pragma unroll
    for (int mt = 0; mt < 2; ++mt) {
      float r0[8];
#pragma unroll
      for (int v = 0; v < 8; ++v) {
        float val = acc[mt * 4 + j][v] * g + bt;
        if (do_relu) val = val > 0.f ? val : 0.f;
        r0[v] = val;
      }
      float* op = out + (((size_t)b0 * C_ + cout) * H_ + h) * W_ + w0 + mt * 16 + off;
      f4 s0 = {r0[0], r0[1], r0[2], r0[3]};
      f4 s1 = {r0[4], r0[5], r0[6], r0[7]};
      *(f4*)(op)     = s0;
      *(f4*)(op + 4) = s1;
    }
  }
}

// ---------------------------------------------------------------------------
// Host-side launch sequence (all on `stream`, stream-ordered dependencies).
// ---------------------------------------------------------------------------
extern "C" void kernel_launch(void* const* d_in, const int* in_sizes, int n_in,
                              void* d_out, int out_size, void* d_ws, size_t ws_size,
                              hipStream_t stream) {
  (void)in_sizes; (void)n_in; (void)out_size; (void)ws_size;

  const float* x    = (const float*)d_in[0];
  const float* w_p1 = (const float*)d_in[1];
  const float* g_p1 = (const float*)d_in[2];
  const float* b_p1 = (const float*)d_in[3];
  const float* w_p2 = (const float*)d_in[4];
  const float* g_p2 = (const float*)d_in[5];
  const float* b_p2 = (const float*)d_in[6];
  const float* w_p  = (const float*)d_in[7];
  const float* g_p  = (const float*)d_in[8];
  const float* b_p  = (const float*)d_in[9];
  const float* w_c1 = (const float*)d_in[10];
  const float* g_c1 = (const float*)d_in[11];
  const float* b_c1 = (const float*)d_in[12];
  const float* w_c2 = (const float*)d_in[13];
  const float* g_c2 = (const float*)d_in[14];
  const float* b_c2 = (const float*)d_in[15];
  float* out = (float*)d_out;

  const size_t XB_BYTES  = (size_t)B_ * H_ * W_ * C_ * 2;     // 37.7 MB
  const size_t WF_BYTES  = (size_t)2304 * 256 * 2;            // 1.18 MB each
  const size_t BUF_BYTES = (size_t)B_ * C_ * H_ * W_ * 4;     // 75.5 MB
  const size_t CM_BYTES  = (size_t)B_ * C_ * W_ * 4;          // 0.79 MB

  char* ws = (char*)d_ws;
  size_t off = 0;
  auto take = [&](size_t n) { char* p = ws + off; off += (n + 255) & ~(size_t)255; return p; };

  unsigned short* XB   = (unsigned short*)take(XB_BYTES);     // x, NHWC bf16
  unsigned short* WFp1 = (unsigned short*)take(WF_BYTES);
  unsigned short* WFp2 = (unsigned short*)take(WF_BYTES);
  unsigned short* WFp  = (unsigned short*)take(WF_BYTES);
  unsigned short* WFc1 = (unsigned short*)take(WF_BYTES);
  unsigned short* WFc2 = (unsigned short*)take(WF_BYTES);
  float*          BUF1 = (float*)take(BUF_BYTES);             // p1 -> p_conv
  float*          BUF2 = (float*)take(BUF_BYTES);             // p2 -> conv1
  unsigned short* SB   = (unsigned short*)take(XB_BYTES);     // s  -> relu(pc+c1)
  float*          CMAX = (float*)take(CM_BYTES);
  float*          RMAX = (float*)take(CM_BYTES);

  // Weight swizzle + activation layout prep.
  k_prep_w<<<2304, 256, 0, stream>>>(w_p1, WFp1);
  k_prep_w<<<2304, 256, 0, stream>>>(w_p2, WFp2);
  k_prep_w<<<2304, 256, 0, stream>>>(w_p,  WFp);
  k_prep_w<<<2304, 256, 0, stream>>>(w_c1, WFc1);
  k_prep_w<<<2304, 256, 0, stream>>>(w_c2, WFc2);
  k_nchw_to_nhwc_bf16<<<B_ * H_, 256, 0, stream>>>(x, x, XB, 0);

  dim3 cgrid(W_ / 32, H_, B_);
  dim3 cblk(128);

  // p1 = relu(bn(conv(x)));  p2 = relu(bn(conv(x)))
  k_conv3x3_wmma<<<cgrid, cblk, 0, stream>>>(XB, WFp1, g_p1, b_p1, BUF1, 1);
  k_conv3x3_wmma<<<cgrid, cblk, 0, stream>>>(XB, WFp2, g_p2, b_p2, BUF2, 1);

  // Corner pools collapse to column/row max broadcast; s = pool1 + pool2.
  k_colmax<<<(B_ * C_ * W_) / 256, 256, 0, stream>>>(BUF1, CMAX);
  k_rowmax<<<(B_ * C_ * H_) / 256, 256, 0, stream>>>(BUF2, RMAX);
  k_make_s<<<(B_ * H_ * W_ * C_) / 256, 256, 0, stream>>>(CMAX, RMAX, SB);

  // conv1 = bn(conv(x))  (BUF2 free after rowmax);  p_conv = bn(conv(s)).
  k_conv3x3_wmma<<<cgrid, cblk, 0, stream>>>(XB, WFc1, g_c1, b_c1, BUF2, 0);
  k_conv3x3_wmma<<<cgrid, cblk, 0, stream>>>(SB, WFp, g_p, b_p, BUF1, 0);

  // rt = relu(p_conv + conv1) -> NHWC bf16 (reuse SB).
  k_nchw_to_nhwc_bf16<<<B_ * H_, 256, 0, stream>>>(BUF1, BUF2, SB, 1);

  // out = relu(bn(conv(rt))), f32 NCHW.
  k_conv3x3_wmma<<<cgrid, cblk, 0, stream>>>(SB, WFc2, g_c2, b_c2, out, 1);
}
